// HWxHW_Attention_Norm_81346680586636
// MI455X (gfx1250) — matrix operand order repaired and checked by reading, not compiled, not run on hardware
//
#include <hip/hip_runtime.h>

typedef __attribute__((ext_vector_type(16))) _Float16 v16h;
typedef __attribute__((ext_vector_type(8)))  _Float16 v8h;
typedef __attribute__((ext_vector_type(8)))  float    v8f;
typedef __attribute__((ext_vector_type(4)))  unsigned su4;
typedef __attribute__((ext_vector_type(8)))  unsigned su8;

#define NTOK 16384
#define CDIM 48
#define CP   64
#define NKB  (NTOK / 64)

__device__ __forceinline__ v8f wmma16(v16h a, v16h b, v8f c) {
  return __builtin_amdgcn_wmma_f32_16x16x32_f16(false, a, false, b, (short)0, c, false, false);
}

// Row-major 16-bit A fragment: two contiguous b128 loads per lane.
__device__ __forceinline__ v16h load_A16(const _Float16* rowbase, int hi) {
  v16h a;
  ((v8h*)&a)[0] = *(const v8h*)(rowbase + hi * 16);
  ((v8h*)&a)[1] = *(const v8h*)(rowbase + hi * 16 + 8);
  return a;
}

// B fragment with K contiguous per lane: two b128 loads.
__device__ __forceinline__ v16h load_B16(const _Float16* base) {
  v16h b;
  ((v8h*)&b)[0] = *(const v8h*)(base);
  ((v8h*)&b)[1] = *(const v8h*)(base + 8);
  return b;
}

// Async contiguous global->LDS 16B copy (per-lane), tracked by ASYNCcnt.
__device__ __forceinline__ void async_copy_b128(const void* gsrc, void* ldst) {
  unsigned lds = (unsigned)(unsigned long long)(uintptr_t)ldst;  // low 32 = LDS offset
  unsigned long long g = (unsigned long long)(uintptr_t)gsrc;
  asm volatile("global_load_async_to_lds_b128 %0, %1, off"
               :: "v"(lds), "v"(g) : "memory");
}

// TDM: 2-D tile load (16-bit elements) via tensor descriptor in SGPRs.
__device__ __forceinline__ void tdm_load_2d(const void* gtile, unsigned lds_off,
                                            unsigned dim0, unsigned dim1,
                                            unsigned tile0, unsigned tile1,
                                            unsigned long long stride0) {
  unsigned long long ga = (unsigned long long)(uintptr_t)gtile;
  su4 g0;
  g0[0] = 1u;                                            // count=1 (user D#)
  g0[1] = lds_off;                                       // lds_addr (bytes)
  g0[2] = (unsigned)(ga & 0xffffffffull);                // global_addr[31:0]
  g0[3] = (unsigned)((ga >> 32) & 0x1ffffffull) | (2u << 30);  // addr[56:32]|type=2
  su8 g1;
  g1[0] = (1u << 16);                                    // data_size = 2 bytes
  g1[1] = (dim0 & 0xffffu) << 16;                        // tensor_dim0[15:0]
  g1[2] = ((dim0 >> 16) & 0xffffu) | ((dim1 & 0xffffu) << 16);
  g1[3] = ((dim1 >> 16) & 0xffffu) | ((tile0 & 0xffffu) << 16);  // tile_dim0
  g1[4] = (tile1 & 0xffffu);                             // tile_dim1, tile_dim2=0
  g1[5] = (unsigned)(stride0 & 0xffffffffull);           // dim0_stride[31:0]
  g1[6] = (unsigned)((stride0 >> 32) & 0xffffu);         // dim0_stride[47:32]
  g1[7] = 0u;
  asm volatile("tensor_load_to_lds %0, %1" :: "s"(g0), "s"(g1) : "memory");
}

// ---------------------------------------------------------------------------
// Kernel 1: qkv = W_qkv (144x48) @ x (48 x 16384)
// ---------------------------------------------------------------------------
__global__ __launch_bounds__(256)
void k_qkv(const float* __restrict__ x, const float* __restrict__ w,
           float* __restrict__ qkv) {
  const int lane  = threadIdx.x & 31;
  const int wv    = threadIdx.x >> 5;
  const int mbase = blockIdx.x * 16;
  const int ptile = blockIdx.y * 8 + wv;
  const int ln15  = lane & 15;
  const int hi    = (lane >> 4) & 1;
  const int row   = mbase + ln15;
  const int ab0   = hi ? 16 : 0;

  v16h a0, a1;
#pragma unroll
  for (int e = 0; e < 16; ++e) {
    int kk = ab0 + e;
    a0[e] = (_Float16)w[row * CDIM + kk];
    int k1 = kk + 32;
    a1[e] = (k1 < CDIM) ? (_Float16)w[row * CDIM + k1] : (_Float16)0.0f;
  }

  const int pix = ptile * 16 + ln15;
  const int kb  = hi ? 16 : 0;
  v16h b0, b1;
#pragma unroll
  for (int e = 0; e < 16; ++e) {
    int k0 = kb + e;
    b0[e] = (_Float16)x[(size_t)k0 * NTOK + pix];
    int k1 = 32 + kb + e;
    b1[e] = (k1 < CDIM) ? (_Float16)x[(size_t)k1 * NTOK + pix] : (_Float16)0.0f;
  }
  v8f c = {};
  c = wmma16(a0, b0, c);
  c = wmma16(a1, b1, c);
#pragma unroll
  for (int r = 0; r < 8; ++r)
    qkv[(size_t)(mbase + r + hi * 8) * NTOK + pix] = c[r];
}

// ---------------------------------------------------------------------------
// Kernel 2: depthwise 3x3 SAME conv, 144 channels of 128x128
// ---------------------------------------------------------------------------
__global__ __launch_bounds__(256)
void k_dw(const float* __restrict__ qkv, const float* __restrict__ dww,
          float* __restrict__ out) {
  int idx = blockIdx.x * 256 + threadIdx.x;
  int ch = idx >> 14;
  int p  = idx & (NTOK - 1);
  int y  = p >> 7, xx = p & 127;
  float acc = 0.f;
#pragma unroll
  for (int dy = 0; dy < 3; ++dy)
#pragma unroll
    for (int dx = 0; dx < 3; ++dx) {
      int yy = y + dy - 1, xc = xx + dx - 1;
      if (yy >= 0 && yy < 128 && xc >= 0 && xc < 128)
        acc += qkv[(size_t)ch * NTOK + yy * 128 + xc] * dww[ch * 9 + dy * 3 + dx];
    }
  out[idx] = acc;
}

// ---------------------------------------------------------------------------
// Kernel 3: per-token L2 norm of q,k (temperature folded into q); emit f16
// Q token-major [N][64], K token-major [N][64], V channel-major [64][N]
// ---------------------------------------------------------------------------
__global__ __launch_bounds__(256)
void k_norm(const float* __restrict__ dwo, const float* __restrict__ temp,
            _Float16* __restrict__ qt, _Float16* __restrict__ kt,
            _Float16* __restrict__ vcm) {
  int tok = blockIdx.x * 256 + threadIdx.x;
  float nq = 0.f, nk = 0.f;
  for (int c = 0; c < CDIM; ++c) {
    float qv = dwo[(size_t)c * NTOK + tok];
    float kv = dwo[(size_t)(CDIM + c) * NTOK + tok];
    nq += qv * qv; nk += kv * kv;
  }
  float qs = temp[0] / fmaxf(sqrtf(nq), 1e-12f);
  float ks = 1.0f   / fmaxf(sqrtf(nk), 1e-12f);
  for (int c = 0; c < CP; ++c) {
    _Float16 qh = (_Float16)0.f, kh = (_Float16)0.f, vh = (_Float16)0.f;
    if (c < CDIM) {
      qh = (_Float16)(dwo[(size_t)c * NTOK + tok] * qs);
      kh = (_Float16)(dwo[(size_t)(CDIM + c) * NTOK + tok] * ks);
      vh = (_Float16)(dwo[(size_t)(2 * CDIM + c) * NTOK + tok]);
    }
    qt[(size_t)tok * CP + c]    = qh;
    kt[(size_t)tok * CP + c]    = kh;
    vcm[(size_t)c * NTOK + tok] = vh;
  }
}

// Stage one 64-key block: K (contiguous, token-major) via async b128;
// V (strided, channel-major) via Tensor Data Mover (wave 0 only).
__device__ __forceinline__ void stage_tiles(const _Float16* kt, const _Float16* vcm,
                                            _Float16* dK, _Float16* dV,
                                            int key0, int tid, int wv) {
  const _Float16* ksrc = kt + (size_t)key0 * CP;
  async_copy_b128(ksrc + (size_t)tid * 8,         dK + tid * 8);
  async_copy_b128(ksrc + (size_t)(tid + 256) * 8, dK + (tid + 256) * 8);
  if (wv == 0)
    tdm_load_2d(vcm + key0, (unsigned)(uintptr_t)dV,
                NTOK, CP, 64, 64, (unsigned long long)NTOK);
}

// ---------------------------------------------------------------------------
// Kernel 4: flash attention, double-buffered K/V staging (TDM + async-LDS)
// 8 waves/block, one 16-query tile per wave.
// LDS tiles: lKt [key][chan] (QK^T B-frags contiguous per lane),
//            lVc [chan][key] (PV  B-frags contiguous per lane).
// ---------------------------------------------------------------------------
__global__ __launch_bounds__(256)
void k_attn(const _Float16* __restrict__ qt, const _Float16* __restrict__ kt,
            const _Float16* __restrict__ vcm, _Float16* __restrict__ O) {
  __shared__ __align__(16) _Float16 lKt[2][64 * CP];   // [buf][key][chan]
  __shared__ __align__(16) _Float16 lVc[2][CP * 64];   // [buf][chan][key]
  __shared__ __align__(16) _Float16 lP[8][16 * 64];    // per-wave P staging
  const int tid  = threadIdx.x;
  const int lane = tid & 31;
  const int wv   = tid >> 5;
  const int ln15 = lane & 15;
  const int hi   = (lane >> 4) & 1;
  const int kbse = hi ? 16 : 0;
  const int qbase = (blockIdx.x * 8 + wv) * 16;

  // Q fragments, held for the whole pass
  const int qrow = qbase + ln15;
  v16h aq0 = load_A16(qt + (size_t)qrow * CP, hi);
  v16h aq1 = load_A16(qt + (size_t)qrow * CP + 32, hi);

  v8f acc0 = {}, acc1 = {}, acc2 = {}, acc3 = {};
  float mrow[8], lrow[8];
#pragma unroll
  for (int r = 0; r < 8; ++r) { mrow[r] = -3.0e38f; lrow[r] = 0.f; }

  // prologue: stage block 0 into buffer 0
  stage_tiles(kt, vcm, lKt[0], lVc[0], 0, tid, wv);

  for (int kb = 0; kb < NKB; ++kb) {
    const int cur = kb & 1;
    // wait for this wave's copies into buf[cur]
    asm volatile("s_wait_asynccnt 0x0" ::: "memory");
    if (wv == 0) __builtin_amdgcn_s_wait_tensorcnt(0);
    __syncthreads();   // buf[cur] visible; all waves done reading buf[1-cur]

    // overlap: stage next block into the other buffer while computing
    if (kb + 1 < NKB)
      stage_tiles(kt, vcm, lKt[1 - cur], lVc[1 - cur], (kb + 1) * 64, tid, wv);

    const _Float16* bK = lKt[cur];
    const _Float16* bV = lVc[cur];

    // S = Q * K^T : 4 key sub-tiles, K-dim 64 -> 2 WMMA each
    float S[4][8];
#pragma unroll
    for (int j = 0; j < 4; ++j) {
      const _Float16* cb = bK + (j * 16 + ln15) * CP + kbse;
      v16h b0 = load_B16(cb);        // channels 0..31 slice
      v16h b1 = load_B16(cb + 32);   // channels 32..63 slice
      v8f s = {};
      s = wmma16(aq0, b0, s);
      s = wmma16(aq1, b1, s);
#pragma unroll
      for (int r = 0; r < 8; ++r) S[j][r] = s[r];
    }

    // online softmax per query row
#pragma unroll
    for (int r = 0; r < 8; ++r) {
      float rm = fmaxf(fmaxf(S[0][r], S[1][r]), fmaxf(S[2][r], S[3][r]));
      rm = fmaxf(rm, __shfl_xor(rm, 1));
      rm = fmaxf(rm, __shfl_xor(rm, 2));
      rm = fmaxf(rm, __shfl_xor(rm, 4));
      rm = fmaxf(rm, __shfl_xor(rm, 8));
      float mnew  = fmaxf(mrow[r], rm);
      float alpha = __expf(mrow[r] - mnew);
      float rs = 0.f;
#pragma unroll
      for (int j = 0; j < 4; ++j) {
        float pv = __expf(S[j][r] - mnew);
        S[j][r] = pv; rs += pv;
      }
      rs += __shfl_xor(rs, 1);
      rs += __shfl_xor(rs, 2);
      rs += __shfl_xor(rs, 4);
      rs += __shfl_xor(rs, 8);
      lrow[r] = lrow[r] * alpha + rs;
      mrow[r] = mnew;
      acc0[r] *= alpha; acc1[r] *= alpha; acc2[r] *= alpha; acc3[r] *= alpha;
    }

    // P: C-layout -> row-major LDS -> A-layout
#pragma unroll
    for (int j = 0; j < 4; ++j)
#pragma unroll
      for (int r = 0; r < 8; ++r)
        lP[wv][(r + hi * 8) * 64 + j * 16 + ln15] = (_Float16)S[j][r];
    asm volatile("s_wait_dscnt 0x0" ::: "memory");

    v16h ap0 = load_A16(&lP[wv][ln15 * 64], hi);
    v16h ap1 = load_A16(&lP[wv][ln15 * 64 + 32], hi);

    // O += P(16x64) * V(64x64) : 4 channel tiles, 2 WMMA each
#pragma unroll
    for (int ct = 0; ct < 4; ++ct) {
      const _Float16* cb = bV + (ct * 16 + ln15) * 64 + kbse;
      v16h b0 = load_B16(cb);        // keys 0..31 slice
      v16h b1 = load_B16(cb + 32);   // keys 32..63 slice
      v8f* accp = (ct == 0) ? &acc0 : (ct == 1) ? &acc1 : (ct == 2) ? &acc2 : &acc3;
      *accp = wmma16(ap0, b0, *accp);
      *accp = wmma16(ap1, b1, *accp);
    }
  }

  // normalize, store O token-major f16 [N][64]
#pragma unroll
  for (int r = 0; r < 8; ++r) {
    float inv = 1.0f / lrow[r];
    int orow = qbase + r + hi * 8;
    O[(size_t)orow * CP +  0 + ln15] = (_Float16)(acc0[r] * inv);
    O[(size_t)orow * CP + 16 + ln15] = (_Float16)(acc1[r] * inv);
    O[(size_t)orow * CP + 32 + ln15] = (_Float16)(acc2[r] * inv);
    O[(size_t)orow * CP + 48 + ln15] = (_Float16)(acc3[r] * inv);
  }
}

// ---------------------------------------------------------------------------
// Kernel 5: out = proj_w (48x48) @ O^T (48 x 16384), f32 out [48][16384]
// ---------------------------------------------------------------------------
__global__ __launch_bounds__(256)
void k_proj(const _Float16* __restrict__ O, const float* __restrict__ pw,
            float* __restrict__ out) {
  const int lane  = threadIdx.x & 31;
  const int wv    = threadIdx.x >> 5;
  const int ln15  = lane & 15;
  const int hi    = (lane >> 4) & 1;
  const int mbase = blockIdx.x * 16;
  const int ptile = blockIdx.y * 8 + wv;
  const int row   = mbase + ln15;
  const int ab0   = hi ? 16 : 0;

  v16h a0, a1;
#pragma unroll
  for (int e = 0; e < 16; ++e) {
    int kk = ab0 + e;
    a0[e] = (_Float16)pw[row * CDIM + kk];
    int k1 = kk + 32;
    a1[e] = (k1 < CDIM) ? (_Float16)pw[row * CDIM + k1] : (_Float16)0.0f;
  }
  const int pix = ptile * 16 + ln15;
  const int kbq = hi ? 16 : 0;
  v16h b0 = load_B16(O + (size_t)pix * CP + kbq);
  v16h b1 = load_B16(O + (size_t)pix * CP + 32 + kbq);

  v8f c = {};
  c = wmma16(a0, b0, c);
  c = wmma16(a1, b1, c);
#pragma unroll
  for (int r = 0; r < 8; ++r)
    out[(size_t)(mbase + r + hi * 8) * NTOK + pix] = c[r];
}

// ---------------------------------------------------------------------------
extern "C" void kernel_launch(void* const* d_in, const int* in_sizes, int n_in,
                              void* d_out, int out_size, void* d_ws, size_t ws_size,
                              hipStream_t stream) {
  (void)in_sizes; (void)n_in; (void)out_size; (void)ws_size;
  const float* x    = (const float*)d_in[0];
  const float* qw   = (const float*)d_in[1];
  const float* dww  = (const float*)d_in[2];
  const float* pw   = (const float*)d_in[3];
  const float* temp = (const float*)d_in[4];

  char* base = (char*)d_ws;
  float*    qkv = (float*)base;                                // 144*N f32
  float*    dwo = (float*)(base + (size_t)144 * NTOK * 4);     // 144*N f32
  _Float16* qt  = (_Float16*)(base + (size_t)288 * NTOK * 4);  // N*64 f16
  _Float16* kt  = qt  + (size_t)NTOK * CP;                     // N*64 f16
  _Float16* vcm = kt  + (size_t)NTOK * CP;                     // 64*N f16
  _Float16* Ob  = vcm + (size_t)NTOK * CP;                     // N*64 f16

  k_qkv <<<dim3(9, 128),           256, 0, stream>>>(x, qw, qkv);
  k_dw  <<<dim3(144 * NTOK / 256), 256, 0, stream>>>(qkv, dww, dwo);
  k_norm<<<dim3(NTOK / 256),       256, 0, stream>>>(dwo, temp, qt, kt, vcm);
  k_attn<<<dim3(NTOK / 128),       256, 0, stream>>>(qt, kt, vcm, Ob);
  k_proj<<<dim3(3, 128),           256, 0, stream>>>(Ob, pw, (float*)d_out);
}
